// InfoNCE_59785944760611
// MI455X (gfx1250) — matrix operand order repaired and checked by reading, not compiled, not run on hardware
//
#include <hip/hip_runtime.h>
#include <hip/hip_bf16.h>
#include <math.h>

#define NS   1024   // samples
#define XD   768    // feature dim (K), 24 * 32
#define LOW  300    // LOWER
#define LOWP 320    // LOWER padded to 20 tiles of 16

typedef __attribute__((ext_vector_type(16))) __bf16 v16bf;
typedef __attribute__((ext_vector_type(8)))  float  v8f;

__device__ __forceinline__ unsigned short f32_to_bf16(float f) {
    union { float f; unsigned int u; } c; c.f = f;
    unsigned int u = c.u;
    unsigned int r = u + 0x7FFFu + ((u >> 16) & 1u);   // round-to-nearest-even
    if ((u & 0x7F800000u) == 0x7F800000u) r = u;        // keep inf/nan bits
    return (unsigned short)(r >> 16);
}

__device__ __forceinline__ float softplus_stable(float x) {
    return fmaxf(x, 0.f) + log1pf(expf(-fabsf(x)));
}

// ---------------- conversion kernels ----------------
__global__ void cvt_f32_bf16(const float* __restrict__ in,
                             unsigned short* __restrict__ out, int n) {
    int i = blockIdx.x * blockDim.x + threadIdx.x;
    if (i < n) out[i] = f32_to_bf16(in[i]);
}

// W1 is (XD+XD) x LOW row-major. Build padded Wx (XD x LOWP) and Wy (XD x LOWP), zero cols >= LOW.
__global__ void cvt_w(const float* __restrict__ W1,
                      unsigned short* __restrict__ Wxb,
                      unsigned short* __restrict__ Wyb) {
    int i = blockIdx.x * blockDim.x + threadIdx.x;      // over XD*LOWP
    if (i >= XD * LOWP) return;
    int k = i / LOWP, n = i % LOWP;
    unsigned short zx = 0, zy = 0;
    if (n < LOW) {
        zx = f32_to_bf16(W1[k * LOW + n]);
        zy = f32_to_bf16(W1[(XD + k) * LOW + n]);
    }
    Wxb[i] = zx;
    Wyb[i] = zy;
}

// ---------------- WMMA GEMM: C(NS x LOWP) = A(NS x XD) * B(XD x LOWP), bf16 in, f32 out ----
// One wave (32 threads) per 16x16 tile. grid = (NS/16, LOWP/16).
__global__ void __launch_bounds__(32)
gemm_bf16_wmma(const unsigned short* __restrict__ A,
               const unsigned short* __restrict__ B,
               float* __restrict__ Crow,    // row-major NS x LOWP (may be null)
               float* __restrict__ Ctr) {   // transposed LOWP x NS (may be null)
    const int mt = blockIdx.x;
    const int nt = blockIdx.y;
    const int lane = threadIdx.x;          // 0..31 (wave32)
    const int half = lane >> 4;            // A K-half select
    const int l15  = lane & 15;

    const unsigned short* arow = A + (mt * 16 + l15) * XD;  // A row for this lane
    const int aoff = half * 8;                              // lanes 16-31 take K+8 / K+24

    v8f c = {};
    #pragma unroll 4
    for (int k0 = 0; k0 < XD; k0 += 32) {
        union { uint4 u[2]; v16bf v; } au, bu;
        // A 16x32 bf16 layout: lane l<16 -> row l, K = k0+[0..7] and k0+[16..23]
        //                      lane l>=16 -> row l-16, K = k0+[8..15] and k0+[24..31]
        au.u[0] = *(const uint4*)(arow + k0 + aoff);
        au.u[1] = *(const uint4*)(arow + k0 + aoff + 16);
        // B 32x16 bf16 layout: lane -> column N = l15, K = k0+aoff+[0..7] and k0+aoff+[16..23]
        const unsigned short* bp = B + (k0 + aoff) * LOWP + nt * 16 + l15;
        #pragma unroll
        for (int s = 0; s < 16; ++s)
            bu.v[s] = __builtin_bit_cast(__bf16, bp[(s < 8 ? s : s + 8) * LOWP]);
        c = __builtin_amdgcn_wmma_f32_16x16x32_bf16(
                false, au.v, false, bu.v, (short)0, c, false, false);
    }

    // C/D layout: VGPR r -> M = r + 8*half, N = l15
    const int n = nt * 16 + l15;
    const int mb = mt * 16 + half * 8;
    #pragma unroll
    for (int r = 0; r < 8; ++r) {
        float v = c[r];
        if (Crow) Crow[(mb + r) * LOWP + n] = v;
        if (Ctr)  Ctr[n * NS + (mb + r)]   = v;
    }
}

// ---------------- accumulator init ----------------
__global__ void init_acc(float* acc) { acc[0] = 0.f; acc[1] = 0.f; }

// ---------------- T0: softplus(relu(hx[i]+hy[i]+b1) . w2 + b2), summed over i ----------
__global__ void __launch_bounds__(256)
t0_kernel(const float* __restrict__ hx, const float* __restrict__ hy,
          const float* __restrict__ b1, const float* __restrict__ W2,
          const float* __restrict__ b2, float* __restrict__ acc) {
    const int i = blockIdx.x;
    __shared__ float red[256];
    float s = 0.f;
    for (int k = threadIdx.x; k < LOW; k += 256) {
        float h = hx[i * LOWP + k] + hy[i * LOWP + k] + b1[k];
        s = fmaf(fmaxf(h, 0.f), W2[k], s);
    }
    red[threadIdx.x] = s;
    __syncthreads();
    for (int off = 128; off > 0; off >>= 1) {
        if (threadIdx.x < off) red[threadIdx.x] += red[threadIdx.x + off];
        __syncthreads();
    }
    if (threadIdx.x == 0)
        atomicAdd(&acc[0], softplus_stable(red[0] + b2[0]));
}

// ---------------- pair term: per row i, LSE_j softplus(relu(hy[i]+hx[j]+b1).w2+b2) -----
// hxT is LOWP x NS (k-major) so the j loop streams coalesced float4 from L2.
__global__ void __launch_bounds__(256)
pair_kernel(const float* __restrict__ hxT, const float* __restrict__ hy,
            const float* __restrict__ b1, const float* __restrict__ W2,
            const float* __restrict__ b2, float* __restrict__ acc) {
    const int i = blockIdx.x;
    const int t = threadIdx.x;                 // 256 threads, 4 j's each
    __shared__ float hyb[LOW];
    __shared__ float w2s[LOW];
    __shared__ float red[256];

    for (int k = t; k < LOW; k += 256) {
        hyb[k] = hy[i * LOWP + k] + b1[k];
        w2s[k] = W2[k];
    }
    __syncthreads();

    float a0 = 0.f, a1 = 0.f, a2 = 0.f, a3 = 0.f;
    const float* hp = hxT + 4 * t;             // j = 4t .. 4t+3
    #pragma unroll 4
    for (int k = 0; k < LOW; ++k) {
        float4 v = *(const float4*)(hp + k * NS);
        float h = hyb[k], w = w2s[k];
        a0 = fmaf(fmaxf(h + v.x, 0.f), w, a0);
        a1 = fmaf(fmaxf(h + v.y, 0.f), w, a1);
        a2 = fmaf(fmaxf(h + v.z, 0.f), w, a2);
        a3 = fmaf(fmaxf(h + v.w, 0.f), w, a3);
    }
    const float bb = b2[0];
    float s0 = softplus_stable(a0 + bb);
    float s1 = softplus_stable(a1 + bb);
    float s2 = softplus_stable(a2 + bb);
    float s3 = softplus_stable(a3 + bb);

    // block max
    red[t] = fmaxf(fmaxf(s0, s1), fmaxf(s2, s3));
    __syncthreads();
    for (int off = 128; off > 0; off >>= 1) {
        if (t < off) red[t] = fmaxf(red[t], red[t + off]);
        __syncthreads();
    }
    const float M = red[0];
    __syncthreads();

    // block sum of exp
    red[t] = expf(s0 - M) + expf(s1 - M) + expf(s2 - M) + expf(s3 - M);
    __syncthreads();
    for (int off = 128; off > 0; off >>= 1) {
        if (t < off) red[t] += red[t + off];
        __syncthreads();
    }
    if (t == 0)
        atomicAdd(&acc[1], M + logf(red[0]));
}

// ---------------- finalize ----------------
__global__ void finalize_kernel(const float* __restrict__ acc, float* __restrict__ out) {
    out[0] = acc[0] * (1.f / NS) - acc[1] * (1.f / NS) - logf((float)NS);
}

extern "C" void kernel_launch(void* const* d_in, const int* in_sizes, int n_in,
                              void* d_out, int out_size, void* d_ws, size_t ws_size,
                              hipStream_t stream) {
    const float* x  = (const float*)d_in[0];
    const float* y  = (const float*)d_in[1];
    const float* W1 = (const float*)d_in[2];
    const float* b1 = (const float*)d_in[3];
    const float* W2 = (const float*)d_in[4];
    const float* b2 = (const float*)d_in[5];
    float* out = (float*)d_out;

    char* ws = (char*)d_ws;
    size_t off = 0;
    auto alloc = [&](size_t bytes) -> void* {
        void* p = ws + off;
        off = (off + bytes + 255) & ~(size_t)255;
        return p;
    };
    unsigned short* xb  = (unsigned short*)alloc((size_t)NS * XD * 2);
    unsigned short* yb  = (unsigned short*)alloc((size_t)NS * XD * 2);
    unsigned short* Wxb = (unsigned short*)alloc((size_t)XD * LOWP * 2);
    unsigned short* Wyb = (unsigned short*)alloc((size_t)XD * LOWP * 2);
    float* hx  = (float*)alloc((size_t)NS * LOWP * 4);
    float* hy  = (float*)alloc((size_t)NS * LOWP * 4);
    float* hxT = (float*)alloc((size_t)LOWP * NS * 4);
    float* acc = (float*)alloc(8);

    // f32 -> bf16
    cvt_f32_bf16<<<(NS * XD + 255) / 256, 256, 0, stream>>>(x, xb, NS * XD);
    cvt_f32_bf16<<<(NS * XD + 255) / 256, 256, 0, stream>>>(y, yb, NS * XD);
    cvt_w<<<(XD * LOWP + 255) / 256, 256, 0, stream>>>(W1, Wxb, Wyb);

    // projections via v_wmma_f32_16x16x32_bf16; hx also stored transposed for the pair pass
    dim3 gg(NS / 16, LOWP / 16);
    gemm_bf16_wmma<<<gg, 32, 0, stream>>>(xb, Wxb, hx, hxT);
    gemm_bf16_wmma<<<gg, 32, 0, stream>>>(yb, Wyb, hy, nullptr);

    init_acc<<<1, 1, 0, stream>>>(acc);
    t0_kernel<<<NS, 256, 0, stream>>>(hx, hy, b1, W2, b2, acc);
    pair_kernel<<<NS, 256, 0, stream>>>(hxT, hy, b1, W2, b2, acc);
    finalize_kernel<<<1, 1, 0, stream>>>(acc, out);
}
